// GCN_59751585022468
// MI455X (gfx1250) — matrix operand order repaired and checked by reading, not compiled, not run on hardware
//
#include <hip/hip_runtime.h>
#include <hip/hip_bf16.h>
#include <math.h>

// ---------------------------------------------------------------------------
// 2-layer GCN for MI455X (gfx1250, wave32, WMMA).
// Dominant cost: streaming x (100000x512 fp32 = 205MB) once through GEMM1.
// GEMMs use v_wmma_f32_16x16x32_bf16 (bf16 A/B, f32 accumulate).
// ---------------------------------------------------------------------------

typedef __attribute__((ext_vector_type(16))) __bf16 v16bf;
typedef __attribute__((ext_vector_type(8)))  float  v8f;

constexpr int FIN  = 512;
constexpr int FHID = 32;
constexpr int FOUT = 16;

// ---------------- init: deg=1, zero both aggregation buffers ----------------
__global__ void init_kernel(float* __restrict__ deg, float* __restrict__ agg1,
                            float* __restrict__ agg2, int n) {
    long i = (long)blockIdx.x * blockDim.x + threadIdx.x;
    long tot = (long)n * FHID;
    if (i < tot)            agg1[i] = 0.0f;
    if (i < (long)n * FOUT) agg2[i] = 0.0f;
    if (i < n)              deg[i]  = 1.0f;   // self-loop baseline
}

// ---------------- degree accumulation: deg[dst] += 1 ------------------------
__global__ void degree_kernel(const int* __restrict__ dst, float* __restrict__ deg, int e) {
    int i = blockIdx.x * blockDim.x + threadIdx.x;
    if (i < e) atomicAdd(&deg[dst[i]], 1.0f);
}

// ---------------- dinv = rsqrt(deg), in place -------------------------------
__global__ void dinv_kernel(float* __restrict__ d, int n) {
    int i = blockIdx.x * blockDim.x + threadIdx.x;
    if (i < n) d[i] = rsqrtf(d[i]);
}

// ---------------- GEMM1: h1[N,32] = x[N,512] @ W1[512,32]  (WMMA bf16) ------
// Per wave: one 16-row x 32-col tile, two f32 accumulators, K-loop of 16 steps.
__global__ __launch_bounds__(256) void gemm1_kernel(const float* __restrict__ x,
                                                    const float* __restrict__ W1,
                                                    float* __restrict__ h1, int n) {
    // Fragment-ordered W1 in LDS: [kstep(16)][ctile(2)][lane(32)][16 bf16] = 32KB
    __shared__ alignas(32) __bf16 ldsB[16 * 2 * 32 * 16];
    for (int e = threadIdx.x; e < 16 * 2 * 32 * 16; e += 256) {
        int j    = e & 15;
        int lane = (e >> 4) & 31;
        int ct   = (e >> 9) & 1;
        int ks   = e >> 10;
        int K    = ks * 32 + ((lane & 16) ? 16 + j : j);
        int Nc   = ct * 16 + (lane & 15);
        ldsB[e] = (__bf16)W1[K * FHID + Nc];
    }
    __syncthreads();

    int wave = threadIdx.x >> 5;
    int lane = threadIdx.x & 31;
    int rowBase = (blockIdx.x * 8 + wave) * 16;
    if (rowBase >= n) return;                 // uniform per wave: EXEC stays all-1s

    int m    = lane & 15;
    int half = lane >> 4;
    const float* xrow = x + (long)(rowBase + m) * FIN;

    v8f c0 = {}; v8f c1 = {};
    for (int ks = 0; ks < FIN / 32; ++ks) {
        int kb = ks * 32 + half * 16;
        __builtin_prefetch(xrow + kb + 32, 0, 3);    // global_prefetch_b8 next step
        v16bf a;
#pragma unroll
        for (int j = 0; j < 8; ++j) {
            a[j]     = (__bf16)xrow[kb + j];
            a[8 + j] = (__bf16)xrow[kb + 8 + j];
        }
        v16bf b0 = *(const v16bf*)&ldsB[((ks * 2 + 0) * 32 + lane) * 16];
        v16bf b1 = *(const v16bf*)&ldsB[((ks * 2 + 1) * 32 + lane) * 16];
        c0 = __builtin_amdgcn_wmma_f32_16x16x32_bf16(false, a, false, b0, (short)0, c0, false, false);
        c1 = __builtin_amdgcn_wmma_f32_16x16x32_bf16(false, a, false, b1, (short)0, c1, false, false);
    }
#pragma unroll
    for (int r = 0; r < 8; ++r) {
        int row = rowBase + r + half * 8;
        h1[(long)row * FHID + m]      = c0[r];
        h1[(long)row * FHID + 16 + m] = c1[r];
    }
}

// ---------------- edge aggregation: agg[dst] += h[src] * dinv_s*dinv_d ------
template <int F>
__global__ void agg_kernel(const float* __restrict__ h, const int* __restrict__ src,
                           const int* __restrict__ dst, const float* __restrict__ dinv,
                           float* __restrict__ agg, long total) {
    long i = (long)blockIdx.x * blockDim.x + threadIdx.x;
    if (i >= total) return;
    int e = (int)(i / F);
    int f = (int)(i - (long)e * F);
    int s = src[e], d = dst[e];
    float coef = dinv[s] * dinv[d];
    atomicAdd(&agg[(long)d * F + f], h[(long)s * F + f] * coef);
}

// ---------------- layer1 epilogue: agg1 = relu(agg1 + h1*dinv^2 + b1) -------
__global__ void selfloop_relu_kernel(float* __restrict__ agg1, const float* __restrict__ h1,
                                     const float* __restrict__ dinv, const float* __restrict__ b1,
                                     long total) {
    long i = (long)blockIdx.x * blockDim.x + threadIdx.x;
    if (i >= total) return;
    int node = (int)(i >> 5);           // FHID == 32
    int f    = (int)(i & 31);
    float di = dinv[node];
    float z  = agg1[i] + h1[i] * di * di + b1[f];
    agg1[i] = z > 0.0f ? z : 0.0f;
}

// ---------------- GEMM2: h2[N,16] = h1p[N,32] @ W2[32,16]  (1 WMMA/tile) ----
__global__ __launch_bounds__(256) void gemm2_kernel(const float* __restrict__ h1p,
                                                    const float* __restrict__ W2,
                                                    float* __restrict__ h2, int n) {
    __shared__ alignas(32) __bf16 ldsB[32 * 16];   // one 32x16 fragment
    for (int e = threadIdx.x; e < 512; e += 256) {
        int j    = e & 15;
        int lane = e >> 4;
        int K    = (lane & 16) ? 16 + j : j;
        int Nc   = lane & 15;
        ldsB[lane * 16 + j] = (__bf16)W2[K * FOUT + Nc];
    }
    __syncthreads();

    int wave = threadIdx.x >> 5;
    int lane = threadIdx.x & 31;
    int rowBase = (blockIdx.x * 8 + wave) * 16;
    if (rowBase >= n) return;

    int m    = lane & 15;
    int half = lane >> 4;
    const float* arow = h1p + (long)(rowBase + m) * FHID;
    int kb = half * 16;

    v16bf a;
#pragma unroll
    for (int j = 0; j < 8; ++j) {
        a[j]     = (__bf16)arow[kb + j];
        a[8 + j] = (__bf16)arow[kb + 8 + j];
    }
    v16bf b = *(const v16bf*)&ldsB[lane * 16];
    v8f c = {};
    c = __builtin_amdgcn_wmma_f32_16x16x32_bf16(false, a, false, b, (short)0, c, false, false);

#pragma unroll
    for (int r = 0; r < 8; ++r) {
        int row = rowBase + r + half * 8;
        h2[(long)row * FOUT + m] = c[r];
    }
}

// ---------------- final: out = log_softmax(agg2 + h2*dinv^2 + b2) -----------
__global__ void final_kernel(const float* __restrict__ agg2, const float* __restrict__ h2,
                             const float* __restrict__ dinv, const float* __restrict__ b2,
                             float* __restrict__ out, int n) {
    int i = blockIdx.x * blockDim.x + threadIdx.x;
    if (i >= n) return;
    float di = dinv[i];
    float dd = di * di;
    float v[FOUT];
    float mx = -INFINITY;
#pragma unroll
    for (int f = 0; f < FOUT; ++f) {
        float z = agg2[(long)i * FOUT + f] + h2[(long)i * FOUT + f] * dd + b2[f];
        v[f] = z;
        mx = fmaxf(mx, z);
    }
    float s = 0.0f;
#pragma unroll
    for (int f = 0; f < FOUT; ++f) s += __expf(v[f] - mx);
    float ls = __logf(s) + mx;
#pragma unroll
    for (int f = 0; f < FOUT; ++f) out[(long)i * FOUT + f] = v[f] - ls;
}

// ---------------------------------------------------------------------------
extern "C" void kernel_launch(void* const* d_in, const int* in_sizes, int n_in,
                              void* d_out, int out_size, void* d_ws, size_t ws_size,
                              hipStream_t stream) {
    const float* x  = (const float*)d_in[0];
    const float* W1 = (const float*)d_in[1];
    const float* b1 = (const float*)d_in[2];
    const float* W2 = (const float*)d_in[3];
    const float* b2 = (const float*)d_in[4];
    const int*   ei = (const int*)d_in[5];

    const int N = in_sizes[0] / FIN;
    const int E = in_sizes[5] / 2;
    const int* src = ei;
    const int* dst = ei + E;

    // Workspace carve-up (floats): dinv[N] | h1[N*32] | agg1[N*32] | h2[N*16] | agg2[N*16]
    float* ws   = (float*)d_ws;
    float* dinv = ws;
    float* h1   = dinv + N;
    float* agg1 = h1   + (size_t)N * FHID;
    float* h2   = agg1 + (size_t)N * FHID;
    float* agg2 = h2   + (size_t)N * FOUT;

    const int T = 256;
    long initTot = (long)N * FHID;
    init_kernel<<<(unsigned)((initTot + T - 1) / T), T, 0, stream>>>(dinv, agg1, agg2, N);
    degree_kernel<<<(unsigned)((E + T - 1) / T), T, 0, stream>>>(dst, dinv, E);
    dinv_kernel<<<(unsigned)((N + T - 1) / T), T, 0, stream>>>(dinv, N);

    int rowTiles = (N + 15) / 16;
    int gemmBlocks = (rowTiles + 7) / 8;          // 8 waves (16-row tiles) per block
    gemm1_kernel<<<gemmBlocks, T, 0, stream>>>(x, W1, h1, N);

    long t32 = (long)E * FHID;
    agg_kernel<FHID><<<(unsigned)((t32 + T - 1) / T), T, 0, stream>>>(h1, src, dst, dinv, agg1, t32);
    selfloop_relu_kernel<<<(unsigned)((initTot + T - 1) / T), T, 0, stream>>>(agg1, h1, dinv, b1, initTot);

    gemm2_kernel<<<gemmBlocks, T, 0, stream>>>(agg1, W2, h2, N);

    long t16 = (long)E * FOUT;
    agg_kernel<FOUT><<<(unsigned)((t16 + T - 1) / T), T, 0, stream>>>(h2, src, dst, dinv, agg2, t16);
    final_kernel<<<(unsigned)((N + T - 1) / T), T, 0, stream>>>(agg2, h2, dinv, b2, (float*)d_out, N);
}
